// NeuralSparse_84524956385437
// MI455X (gfx1250) — compile-verified, matrix operand order checked by reading, not run on hardware
//
#include <hip/hip_runtime.h>
#include <hip/hip_bf16.h>

#define NN   50000
#define FF   128
#define HH   128
#define EE   800000
#define KK   5
#define TAUF 0.9991f

typedef __attribute__((ext_vector_type(16))) __bf16        v16bf;
typedef __attribute__((ext_vector_type(8)))  float         v8f;
typedef __attribute__((ext_vector_type(8)))  unsigned int  v8u;

__device__ __forceinline__ unsigned short f2bf(float f) {
    unsigned u = __float_as_uint(f);
    u += 0x7FFFu + ((u >> 16) & 1u);          // round-to-nearest-even
    return (unsigned short)(u >> 16);
}
// order-preserving float <-> uint encoding for atomicMax on floats
__device__ __forceinline__ unsigned encf(float f) {
    unsigned u = __float_as_uint(f);
    return (u & 0x80000000u) ? ~u : (u | 0x80000000u);
}
__device__ __forceinline__ float decf(unsigned e) {
    return __uint_as_float((e & 0x80000000u) ? (e ^ 0x80000000u) : ~e);
}

// ---------------- prep: fp32 -> bf16 node features ----------------
__global__ void k_cvt(const float* __restrict__ x, unsigned short* __restrict__ xb, int n) {
    int i = blockIdx.x * blockDim.x + threadIdx.x;
    if (i < n) xb[i] = f2bf(x[i]);
}

// ---------------- prep: W1 (256x128 f32) -> WMMA B fragments ----------------
// layout: [frag f=k*8+t][lane][r 0..7] packed u32 (two bf16: K0, K0+1)
__global__ void k_prepw(const float* __restrict__ W1, unsigned int* __restrict__ bfrag) {
    int idx = blockIdx.x * blockDim.x + threadIdx.x;
    if (idx >= 64 * 32 * 8) return;
    int f    = idx >> 8;            // 0..63
    int lane = (idx >> 3) & 31;
    int r    = idx & 7;
    int k = f >> 3, t = f & 7;
    int ncol = lane & 15;
    int kbh  = (lane & 16) ? 8 : 0;
    int K0 = k * 32 + ((r >> 2) << 4) + kbh + ((r & 3) << 1);
    int gn = t * 16 + ncol;
    unsigned lo = f2bf(W1[(size_t)K0 * HH + gn]);
    unsigned hi = f2bf(W1[(size_t)(K0 + 1) * HH + gn]);
    bfrag[idx] = lo | (hi << 16);
}

// ---------------- main WMMA edge-MLP: z[e] for all edges ----------------
__global__ void __launch_bounds__(128)
k_mlp(const unsigned short* __restrict__ xbf, const unsigned int* __restrict__ bfrag,
      const float* __restrict__ b1, const float* __restrict__ W2, const float* __restrict__ b2,
      const int* __restrict__ src, const int* __restrict__ dst, float* __restrict__ z) {
    __shared__ __align__(16) unsigned short At[4][16][256];   // 32 KB: 4 waves x 16 rows x concat(2F)
    const int lane = threadIdx.x & 31;
    const int wid  = threadIdx.x >> 5;
    const int eBase = (blockIdx.x * 4 + wid) * 16;            // E % 64 == 0, no tail

    // stage 16 concatenated rows; lanes 0-15: src half, lanes 16-31: dst half (16B chunks)
    #pragma unroll 4
    for (int c = 0; c < 16; ++c) {
        int e = eBase + c;
        int node = (lane < 16) ? src[e] : dst[e];
        int chunk = lane & 15;
        uint4 v = ((const uint4*)(xbf + (size_t)node * FF))[chunk];
        *((uint4*)(&At[wid][c][(lane >> 4) * 128]) + chunk) = v;
    }
    __syncthreads();

    v8f acc[8];
    v8f zero = {0.f, 0.f, 0.f, 0.f, 0.f, 0.f, 0.f, 0.f};
    #pragma unroll
    for (int t = 0; t < 8; ++t) acc[t] = zero;

    const int M  = lane & 15;
    const int kb = (lane < 16) ? 0 : 8;
    const unsigned short* arow = &At[wid][M][0];

    #pragma unroll
    for (int k = 0; k < 8; ++k) {
        v8u au;
        #pragma unroll
        for (int r = 0; r < 8; ++r) {
            int K0 = k * 32 + ((r >> 2) << 4) + kb + ((r & 3) << 1);
            au[r] = *(const unsigned int*)(arow + K0);        // ds_load_b32, 4B aligned
        }
        v16bf A = __builtin_bit_cast(v16bf, au);
        #pragma unroll
        for (int t = 0; t < 8; ++t) {
            int f = k * 8 + t;
            const uint4* bp = (const uint4*)bfrag + ((size_t)(f * 32 + lane) * 2);
            uint4 blo = bp[0], bhi = bp[1];
            v8u bu = {blo.x, blo.y, blo.z, blo.w, bhi.x, bhi.y, bhi.z, bhi.w};
            v16bf B = __builtin_bit_cast(v16bf, bu);
            acc[t] = __builtin_amdgcn_wmma_f32_16x16x32_bf16(
                false, A, false, B, (short)0, acc[t], false, false);
        }
    }

    // epilogue: relu(acc + b1) . W2 ; D layout: lanes 0-15 -> M=j, lanes 16-31 -> M=j+8
    const int n = lane & 15;
    float zp[8];
    #pragma unroll
    for (int j = 0; j < 8; ++j) zp[j] = 0.f;
    #pragma unroll
    for (int t = 0; t < 8; ++t) {
        int col = t * 16 + n;
        float bb = b1[col], w = W2[col];
        #pragma unroll
        for (int j = 0; j < 8; ++j) {
            float h = acc[t][j] + bb;
            zp[j] += (h > 0.f ? h : 0.f) * w;
        }
    }
    #pragma unroll
    for (int j = 0; j < 8; ++j) {                 // reduce across 16 lanes per half-wave
        float v = zp[j];
        v += __shfl_xor(v, 8, 16);
        v += __shfl_xor(v, 4, 16);
        v += __shfl_xor(v, 2, 16);
        v += __shfl_xor(v, 1, 16);
        zp[j] = v;
    }
    if (n == 0) {
        float bias2 = b2[0];
        int mbase = (lane < 16) ? 0 : 8;
        #pragma unroll
        for (int j = 0; j < 8; ++j) z[eBase + mbase + j] = zp[j] + bias2;
    }
}

// ---------------- segmented softmax / gumbel rounds ----------------
__global__ void k_init_node(unsigned* maxU, float* den) {
    int i = blockIdx.x * blockDim.x + threadIdx.x;
    if (i < NN) { maxU[i] = 0u; den[i] = 0.f; }
}
__global__ void k_init_round(unsigned* mU, float* den2, int* sel) {
    int i = blockIdx.x * blockDim.x + threadIdx.x;
    if (i < NN) { mU[i] = 0u; den2[i] = 0.f; sel[i] = (int)0x80000000; }
}
__global__ void k_segmax(const float* __restrict__ z, const int* __restrict__ src, unsigned* maxU) {
    int i = blockIdx.x * blockDim.x + threadIdx.x;
    if (i < EE) atomicMax(&maxU[src[i]], encf(z[i]));
}
__global__ void k_segsum(const float* __restrict__ z, const int* __restrict__ src,
                         const unsigned* __restrict__ maxU, float* den) {
    int i = blockIdx.x * blockDim.x + threadIdx.x;
    if (i < EE) {
        int sn = src[i];
        atomicAdd(&den[sn], __expf(z[i] - decf(maxU[sn])));
    }
}
__global__ void k_logpi(const int* __restrict__ src, const unsigned* __restrict__ maxU,
                        const float* __restrict__ den, float* z) {   // in-place: z -> log_pi
    int i = blockIdx.x * blockDim.x + threadIdx.x;
    if (i < EE) {
        int sn = src[i];
        z[i] = z[i] - decf(maxU[sn]) - __logf(den[sn]);
    }
}
__global__ void k_round1(const float* __restrict__ lp, const float* __restrict__ u,
                         const int* __restrict__ src, float* __restrict__ s, unsigned* mU) {
    int i = blockIdx.x * blockDim.x + threadIdx.x;
    if (i < EE) {
        float g = -__logf(-__logf(u[i]));
        float sc = (lp[i] + g) * (1.0f / TAUF);
        s[i] = sc;
        atomicMax(&mU[src[i]], encf(sc));
    }
}
__global__ void k_round2(const float* __restrict__ s, const int* __restrict__ src,
                         const int* __restrict__ dst, const unsigned* __restrict__ mU,
                         float* den2, int* sel) {
    int i = blockIdx.x * blockDim.x + threadIdx.x;
    if (i < EE) {
        int sn = src[i];
        float m = decf(mU[sn]);
        atomicAdd(&den2[sn], __expf(s[i] - m));
        if (s[i] >= m) atomicMax(&sel[sn], dst[i]);
    }
}
__global__ void k_round3(const float* __restrict__ s, const int* __restrict__ src,
                         const unsigned* __restrict__ mU, const float* __restrict__ den2,
                         float* __restrict__ soft) {
    int i = blockIdx.x * blockDim.x + threadIdx.x;
    if (i < EE) {
        int sn = src[i];
        soft[i] = __expf(s[i] - decf(mU[sn])) / den2[sn];
    }
}
__global__ void k_selwrite(const int* __restrict__ sel, float* __restrict__ out) {
    int i = blockIdx.x * blockDim.x + threadIdx.x;
    if (i < NN) out[i] = (float)sel[i];
}

extern "C" void kernel_launch(void* const* d_in, const int* in_sizes, int n_in,
                              void* d_out, int out_size, void* d_ws, size_t ws_size,
                              hipStream_t stream) {
    const float* x      = (const float*)d_in[0];
    const float* W1     = (const float*)d_in[1];
    const float* b1     = (const float*)d_in[2];
    const float* W2     = (const float*)d_in[3];
    const float* b2     = (const float*)d_in[4];
    const float* unoise = (const float*)d_in[5];
    const int*   src    = (const int*)d_in[6];
    const int*   dst    = (const int*)d_in[7];
    float* out     = (float*)d_out;
    float* selOut  = out;                         // (K, N)
    float* softOut = out + (size_t)KK * NN;       // (K, E)

    char* w = (char*)d_ws;
    auto alloc = [&](size_t bytes) { char* p = w; w += (bytes + 255) & ~(size_t)255; return p; };
    unsigned short* xbf   = (unsigned short*)alloc((size_t)NN * FF * 2);
    unsigned int*   bfrag = (unsigned int*)  alloc((size_t)64 * 32 * 8 * 4);
    float*          z     = (float*)         alloc((size_t)EE * 4);   // z then log_pi (in place)
    float*          s     = (float*)         alloc((size_t)EE * 4);
    unsigned*       maxU  = (unsigned*)      alloc((size_t)NN * 4);
    float*          den   = (float*)         alloc((size_t)NN * 4);
    unsigned*       mU    = (unsigned*)      alloc((size_t)NN * 4);
    float*          den2  = (float*)         alloc((size_t)NN * 4);
    int*            sel   = (int*)           alloc((size_t)NN * 4);

    k_cvt      <<<(NN * FF + 255) / 256, 256, 0, stream>>>(x, xbf, NN * FF);
    k_prepw    <<<(64 * 32 * 8 + 255) / 256, 256, 0, stream>>>(W1, bfrag);
    k_init_node<<<(NN + 255) / 256, 256, 0, stream>>>(maxU, den);

    k_mlp      <<<EE / 64, 128, 0, stream>>>(xbf, bfrag, b1, W2, b2, src, dst, z);

    k_segmax   <<<EE / 256, 256, 0, stream>>>(z, src, maxU);
    k_segsum   <<<EE / 256, 256, 0, stream>>>(z, src, maxU, den);
    k_logpi    <<<EE / 256, 256, 0, stream>>>(src, maxU, den, z);

    for (int k = 0; k < KK; ++k) {
        k_init_round<<<(NN + 255) / 256, 256, 0, stream>>>(mU, den2, sel);
        k_round1    <<<EE / 256, 256, 0, stream>>>(z, unoise + (size_t)k * EE, src, s, mU);
        k_round2    <<<EE / 256, 256, 0, stream>>>(s, src, dst, mU, den2, sel);
        k_round3    <<<EE / 256, 256, 0, stream>>>(s, src, mU, den2, softOut + (size_t)k * EE);
        k_selwrite  <<<(NN + 255) / 256, 256, 0, stream>>>(sel, selOut + (size_t)k * NN);
    }
}
